// MinEuclideanDistBlock_48825188221038
// MI455X (gfx1250) — matrix operand order repaired and checked by reading, not compiled, not run on hardware
//
#include <hip/hip_runtime.h>
#include <math.h>

typedef __attribute__((ext_vector_type(16))) _Float16 v16h;
typedef __attribute__((ext_vector_type(8)))  float    v8f;
typedef __attribute__((ext_vector_type(4)))  unsigned v4u;

#define B_ 128
#define C_ 3
#define L_ 2048
#define K_ 128
#define S_ 50
#define SP 64                          // S padded to WMMA K multiple
#define W_ (L_ - S_ + 1)               // 1999
#define WT 128                         // windows per block
#define NCHUNK ((W_ + WT - 1) / WT)    // 16
#define XSTRIDE 256                    // padded x-slice stride per channel

union AFrag { v16h h; v4u q[2]; };

__global__ __launch_bounds__(256) void init_out_kernel(unsigned* __restrict__ out) {
    int i = blockIdx.x * 256 + threadIdx.x;
    if (i < B_ * K_) out[i] = 0x7F800000u;   // +inf
}

__global__ __launch_bounds__(256) void shapelet_min_kernel(
    const float* __restrict__ x,     // (B, C, L)
    const float* __restrict__ shp,   // (C, K, S)
    unsigned*    __restrict__ out)   // (B, K) as f32 bits, pre-set to +inf
{
    // LDS declaration order matters: hottest arrays first so their byte
    // offsets fit the 16-bit DS immediate-offset field (no VALU address adds).
    // x slice hi/lo split replicated at 8 shift-parities -> every A-fragment
    // chunk is a 16B-aligned ds_load_b128 (and broadcast-friendly).
    __shared__ __align__(16) _Float16 xhiC[8][C_ * XSTRIDE];  // 12 KB @ 0x0000
    __shared__ __align__(16) _Float16 xloC[8][C_ * XSTRIDE];  // 12 KB @ 0x3000
    __shared__ __align__(32) float sqw[C_ * WT];              //  1.5 KB
    __shared__ float sqs[C_ * K_];                            //  1.5 KB
    __shared__ float xs[C_ * XSTRIDE];                        //  3 KB
    // Shapelets, pre-split into f16 hi/lo, S padded to 64 with zeros (cold-ish:
    // only 4 loads per (c,ks) iteration, so high offsets are fine).
    __shared__ __align__(32) _Float16 sh_hi[C_ * K_ * SP];    // 48 KB
    __shared__ __align__(32) _Float16 sh_lo[C_ * K_ * SP];    // 48 KB

    const int tid   = threadIdx.x;
    const int b     = blockIdx.x / NCHUNK;
    const int chunk = blockIdx.x % NCHUNK;
    const int w0    = chunk * WT;

    // ---- stage x slice: split once into hi/lo, 8 shifted copies ----
    for (int i = tid; i < C_ * XSTRIDE; i += 256) {
        int c = i >> 8, j = i & (XSTRIDE - 1);
        int gi = w0 + j;
        float f = (gi < L_) ? x[((size_t)b * C_ + c) * L_ + gi] : 0.0f;
        _Float16 h = (_Float16)f;
        _Float16 l = (_Float16)(f - (float)h);
        xs[i] = f;
        #pragma unroll
        for (int p = 0; p < 8; ++p) {            // copy p: [p][j-p] = value(j)
            if (j >= p) { xhiC[p][i - p] = h; xloC[p][i - p] = l; }
        }
    }
    // ---- stage shapelets, split once, zero-padded to SP ----
    for (int i = tid; i < C_ * K_ * SP; i += 256) {
        int s = i & (SP - 1), ck = i >> 6;
        float f = (s < S_) ? shp[(size_t)ck * S_ + s] : 0.0f;
        _Float16 h = (_Float16)f;
        sh_hi[i] = h;
        sh_lo[i] = (_Float16)(f - (float)h);
    }
    __syncthreads();

    // ---- squared norms ----
    for (int i = tid; i < C_ * WT; i += 256) {
        int c = i >> 7, wl = i & (WT - 1);
        float s = 0.0f;
        for (int t = 0; t < S_; ++t) { float v = xs[c * XSTRIDE + wl + t]; s += v * v; }
        sqw[i] = s;
    }
    for (int i = tid; i < C_ * K_; i += 256) {
        const float* row = shp + (size_t)i * S_;
        float s = 0.0f;
        for (int t = 0; t < S_; ++t) { float v = row[t]; s += v * v; }
        sqs[i] = s;
    }
    __syncthreads();

    const int lane = tid & 31;
    const int kt   = tid >> 5;     // wave id == k-tile (8 tiles of 16 -> K=128)
    const int n    = lane & 15;    // N index (k within tile)
    const int lh   = lane >> 4;    // lane half
    const int m    = lane & 15;    // M index (w within tile) for A fragment

    // per-lane shift-parity copy and aligned lane offset (uniform for all frags)
    const _Float16* phi8 = xhiC[m & 7];
    const _Float16* plo8 = xloC[m & 7];
    const int moff = (m & ~7) + (lh << 4);

    const v8f vz = {0.f, 0.f, 0.f, 0.f, 0.f, 0.f, 0.f, 0.f};
    float mn = __builtin_inff();

    // two groups of 4 w-tiles: keeps acc/sum register pressure at 64 VGPRs
    for (int g = 0; g < 2; ++g) {
        const int wg = g * 64;
        v8f sum[4];
        #pragma unroll
        for (int mt = 0; mt < 4; ++mt) sum[mt] = vz;

        for (int c = 0; c < C_; ++c) {
            v8f acc[4];
            #pragma unroll
            for (int mt = 0; mt < 4; ++mt) acc[mt] = vz;

            const int   kglob = kt * 16 + n;
            const float sqs_v = sqs[c * K_ + kglob];

            #pragma unroll
            for (int ks = 0; ks < SP; ks += 32) {
                // ---- B fragment: 16 contiguous pre-split halves, 32B aligned ----
                const int boff = (c * K_ + kglob) * SP + ks + (lh << 4);
                v16h Bhi = *(const v16h*)&sh_hi[boff];
                v16h Blo = *(const v16h*)&sh_lo[boff];

                #pragma unroll
                for (int mt = 0; mt < 4; ++mt) {
                    // ---- A fragment: 2x ds_load_b128 per matrix, all aligned ----
                    const int ib = c * XSTRIDE + wg + mt * 16 + ks + moff;
                    AFrag ahi, alo;
                    ahi.q[0] = *(const v4u*)&phi8[ib];
                    ahi.q[1] = *(const v4u*)&phi8[ib + 8];
                    alo.q[0] = *(const v4u*)&plo8[ib];
                    alo.q[1] = *(const v4u*)&plo8[ib + 8];
                    // compensated f32-accuracy dot: hi*hi + hi*lo + lo*hi
                    acc[mt] = __builtin_amdgcn_wmma_f32_16x16x32_f16(
                        false, ahi.h, false, Bhi, (short)0, acc[mt], false, false);
                    acc[mt] = __builtin_amdgcn_wmma_f32_16x16x32_f16(
                        false, ahi.h, false, Blo, (short)0, acc[mt], false, false);
                    acc[mt] = __builtin_amdgcn_wmma_f32_16x16x32_f16(
                        false, alo.h, false, Bhi, (short)0, acc[mt], false, false);
                }
            }

            // ---- finish channel: dist = sqrt(max(||w||^2+||s||^2-2*dot, 0)) ----
            #pragma unroll
            for (int mt = 0; mt < 4; ++mt) {
                v8f sw = *(const v8f*)&sqw[c * WT + wg + mt * 16 + lh * 8];
                v8f d2 = sw + sqs_v - 2.0f * acc[mt];
                #pragma unroll
                for (int r = 0; r < 8; ++r)
                    sum[mt][r] += sqrtf(fmaxf(d2[r], 0.0f));
            }
        }

        // ---- fold this group's windows into the running min ----
        #pragma unroll
        for (int mt = 0; mt < 4; ++mt) {
            #pragma unroll
            for (int r = 0; r < 8; ++r) {
                int wl  = wg + mt * 16 + lh * 8 + r;
                float v = (w0 + wl < W_) ? sum[mt][r] : __builtin_inff();
                mn = fminf(mn, v);
            }
        }
    }

    // lanes l and l+16 hold the same k, complementary window rows
    mn = fminf(mn, __shfl_xor(mn, 16, 32));

    // non-negative floats: u32 bit-pattern min == float min (out pre-set to +inf)
    if (lane < 16)
        atomicMin(&out[(size_t)b * K_ + kt * 16 + lane], __float_as_uint(mn));
}

extern "C" void kernel_launch(void* const* d_in, const int* in_sizes, int n_in,
                              void* d_out, int out_size, void* d_ws, size_t ws_size,
                              hipStream_t stream) {
    const float* x   = (const float*)d_in[0];   // (128, 3, 2048) f32
    const float* shp = (const float*)d_in[1];   // (3, 128, 50)  f32
    unsigned*    out = (unsigned*)d_out;        // (128, 1, 128) f32 bits

    init_out_kernel<<<(B_ * K_ + 255) / 256, 256, 0, stream>>>(out);
    shapelet_min_kernel<<<B_ * NCHUNK, 256, 0, stream>>>(x, shp, out);
}
